// GNDeep_7834020348714
// MI455X (gfx1250) — matrix operand order repaired and checked by reading, not compiled, not run on hardware
//
#include <hip/hip_runtime.h>

typedef __bf16 bf16;
typedef __attribute__((ext_vector_type(16))) __bf16 v16bf;
typedef __attribute__((ext_vector_type(8)))  __bf16 v8bf;
typedef __attribute__((ext_vector_type(8)))  float  v8f;

#define NN 768
#define DD 128

__device__ inline v8f vzero() {
    v8f z;
#pragma unroll
    for (int i = 0; i < 8; ++i) z[i] = 0.0f;
    return z;
}

__device__ inline v8f wmma_bf16(v16bf a, v16bf b, v8f c) {
    return __builtin_amdgcn_wmma_f32_16x16x32_bf16(
        /*neg_a=*/false, a, /*neg_b=*/false, b,
        /*c_mod=*/(short)0, c, /*reuse_a=*/false, /*reuse_b=*/false);
}

__device__ inline v16bf make_frag(const bf16* lo, const bf16* hi) {
    v8bf b0 = *(const v8bf*)lo;
    v8bf b1 = *(const v8bf*)hi;
    v16bf t;
#pragma unroll
    for (int q = 0; q < 8; ++q) { t[q] = b0[q]; t[q + 8] = b1[q]; }
    return t;
}

// ---------------------------------------------------------------------------
// Generic C[M,128] = (beta?C:0) + A[M,128](bf16) @ W[128,128](bf16), f32 out.
// Block = 256 threads = 8 waves; each wave computes 16 rows x 128 cols.
// B fragments for a K-step are batch-loaded so the 8 WMMAs issue back-to-back.
// ---------------------------------------------------------------------------
__global__ __launch_bounds__(256) void gemm128_kernel(
    const bf16* __restrict__ A, const bf16* __restrict__ W,
    float* __restrict__ C, int beta)
{
    __shared__ bf16 ldsWt[DD * DD];  // W transposed: Wt[n][k]
    int tid = threadIdx.x;
    for (int idx = tid; idx < DD * DD; idx += 256) {
        int k = idx >> 7, n = idx & 127;
        ldsWt[n * DD + k] = W[idx];
    }
    __syncthreads();

    int wave = tid >> 5, lane = tid & 31;
    int lmod = lane & 15, lhalf = lane >> 4;
    int row0 = blockIdx.x * 128 + wave * 16;
    int arow = row0 + lmod;

    v8f acc[8];
#pragma unroll
    for (int nt = 0; nt < 8; ++nt) acc[nt] = vzero();

    const bf16* abase = A + (size_t)arow * DD + lhalf * 8;
#pragma unroll
    for (int kk = 0; kk < 4; ++kk) {
        v16bf bfr[8];
#pragma unroll
        for (int nt = 0; nt < 8; ++nt) {
            const bf16* bp = &ldsWt[(nt * 16 + lmod) * DD + kk * 32 + lhalf * 8];
            bfr[nt] = make_frag(bp, bp + 16);
        }
        v16bf af = make_frag(abase + kk * 32, abase + kk * 32 + 16);
#pragma unroll
        for (int nt = 0; nt < 8; ++nt)
            acc[nt] = wmma_bf16(af, bfr[nt], acc[nt]);
    }
#pragma unroll
    for (int p = 0; p < 8; ++p) {
        int grow = row0 + p + lhalf * 8;
        float* cp = C + (size_t)grow * DD + lmod;
#pragma unroll
        for (int nt = 0; nt < 8; ++nt) {
            float v = acc[nt][p];
            if (beta) v += cp[nt * 16];
            cp[nt * 16] = v;
        }
    }
}

// ---------------------------------------------------------------------------
// Edge update kernel:
//   e'[i][j][:] = relu( e[i][j][:] @ W  + se[i][:] + re[j][:] + uterm[:] )
//   S[j][:]    += sum_i e'[i][j][:]        (f32 global atomics, once/block)
// GEN_A:   synthesize e[i][j][c] = relu(s[i][c] + r'[j][c]) on the fly.
// STORE_E: write e' back (bf16) via LDS reassembly for b128 stores.
// The full 128x128 W (32 B-fragments) is hoisted into registers before the
// sender loop: the inner loop is pure A-load + 32 back-to-back WMMAs.
// Grid: (6 j-tiles of 128, 48 i-chunks of 16). Block 256 = 8 waves.
// ---------------------------------------------------------------------------
template <bool GEN_A, bool STORE_E>
__global__ __launch_bounds__(256) void edge_kernel(
    const bf16* __restrict__ Ein, bf16* __restrict__ Eout,
    const float* __restrict__ s_enc, const float* __restrict__ rp_enc,
    const bf16* __restrict__ W,
    const float* __restrict__ se, const float* __restrict__ re,
    const float* __restrict__ uterm,
    float* __restrict__ S)
{
    __shared__ bf16 ldsWt[DD * DD];       // 32 KB
    __shared__ bf16 ldsOut[8 * 16 * DD];  // 32 KB staging (used if STORE_E)

    int tid = threadIdx.x;
    for (int idx = tid; idx < DD * DD; idx += 256) {
        int k = idx >> 7, n = idx & 127;
        ldsWt[n * DD + k] = W[idx];
    }
    __syncthreads();

    int wave = tid >> 5, lane = tid & 31;
    int lmod = lane & 15, lhalf = lane >> 4;
    int jb = blockIdx.x * 128;
    int jrow0 = jb + wave * 16;
    int aj = jrow0 + lmod;            // this lane's A-matrix row (a j index)
    int i0 = blockIdx.y * 16;

    // Hoist the entire weight matrix into register fragments (loop-invariant).
    v16bf bfr[32];
#pragma unroll
    for (int kk = 0; kk < 4; ++kk)
#pragma unroll
        for (int nt = 0; nt < 8; ++nt) {
            const bf16* bp = &ldsWt[(nt * 16 + lmod) * DD + kk * 32 + lhalf * 8];
            bfr[kk * 8 + nt] = make_frag(bp, bp + 16);
        }

    // Per-output-element constants: pre[nt][p] = re[j][c] + uterm[c]
    v8f pre[8], csum[8];
#pragma unroll
    for (int nt = 0; nt < 8; ++nt) {
        int c = nt * 16 + lmod;
        float ut = uterm[c];
        v8f t;
#pragma unroll
        for (int p = 0; p < 8; ++p) {
            int j = jrow0 + p + lhalf * 8;
            t[p] = re[(size_t)j * DD + c] + ut;
        }
        pre[nt] = t;
        csum[nt] = vzero();
    }

    const float* rpl = GEN_A ? (rp_enc + (size_t)aj * DD) : nullptr;

    for (int ii = 0; ii < 16; ++ii) {
        int i = i0 + ii;
        v8f acc[8];
#pragma unroll
        for (int nt = 0; nt < 8; ++nt) acc[nt] = vzero();

        const float* sl = GEN_A ? (s_enc + (size_t)i * DD) : nullptr;
        const bf16* el = GEN_A ? nullptr
                               : (Ein + ((size_t)i * NN + aj) * DD + lhalf * 8);
        if (!GEN_A && ii + 1 < 16) {
            // pull next sender's rows toward the WGP cache
            __builtin_prefetch(el + (size_t)NN * DD, 0, 0);
        }
#pragma unroll
        for (int kk = 0; kk < 4; ++kk) {
            v16bf af;
            if (GEN_A) {
                int cb = kk * 32 + lhalf * 8;
#pragma unroll
                for (int t = 0; t < 8; ++t) {
                    float v0 = sl[cb + t] + rpl[cb + t];
                    float v1 = sl[cb + 16 + t] + rpl[cb + 16 + t];
                    af[t]     = (bf16)fmaxf(v0, 0.0f);
                    af[t + 8] = (bf16)fmaxf(v1, 0.0f);
                }
            } else {
                af = make_frag(el + kk * 32, el + kk * 32 + 16);
            }
#pragma unroll
            for (int nt = 0; nt < 8; ++nt)
                acc[nt] = wmma_bf16(af, bfr[kk * 8 + nt], acc[nt]);
        }

        // Epilogue: bias terms + relu, accumulate column sums, optional store.
#pragma unroll
        for (int nt = 0; nt < 8; ++nt) {
            int c = nt * 16 + lmod;
            float sev = se[(size_t)i * DD + c];
            v8f vals;
#pragma unroll
            for (int p = 0; p < 8; ++p) {
                float v = acc[nt][p] + pre[nt][p] + sev;
                v = fmaxf(v, 0.0f);
                vals[p] = v;
            }
            csum[nt] += vals;
            if (STORE_E) {
#pragma unroll
                for (int p = 0; p < 8; ++p) {
                    int rloc = p + lhalf * 8;
                    ldsOut[wave * 2048 + rloc * DD + c] = (bf16)vals[p];
                }
            }
        }
        if (STORE_E) {
            // LDS ops of a wave are in-order; wait for own writes then copy
            // the wave's 16x128 bf16 tile out with contiguous b128 stores.
            asm volatile("s_wait_dscnt 0" ::: "memory");
            int r = lane >> 1, h = lane & 1;
            const bf16* src = ldsOut + wave * 2048 + r * DD + h * 64;
            bf16* dst = Eout + ((size_t)i * NN + jrow0 + r) * DD + h * 64;
#pragma unroll
            for (int q = 0; q < 8; ++q)
                *(v8bf*)(dst + q * 8) = *(const v8bf*)(src + q * 8);
        }
    }

    // One set of f32 atomics per block for the sender-sum aggregation.
#pragma unroll
    for (int nt = 0; nt < 8; ++nt) {
        int c = nt * 16 + lmod;
#pragma unroll
        for (int p = 0; p < 8; ++p) {
            int j = jrow0 + p + lhalf * 8;
            atomicAdd(&S[(size_t)j * DD + c], csum[nt][p]);
        }
    }
}

// ---------------------------------------------------------------------------
// Small helper kernels
// ---------------------------------------------------------------------------
__global__ void f32_to_bf16_kernel(const float* __restrict__ src,
                                   bf16* __restrict__ dst, int n, float scale) {
    int i = blockIdx.x * 256 + threadIdx.x;
    if (i < n) dst[i] = (bf16)(src[i] * scale);
}

__global__ void add_rowvec_kernel(float* __restrict__ X,
                                  const float* __restrict__ b, int n) {
    int i = blockIdx.x * 256 + threadIdx.x;
    if (i < n) X[i] += b[i & 127];
}

// esum[c] += sum_{i=blockIdx.x, all j} relu(s[i][c] + r'[j][c])
__global__ __launch_bounds__(256) void enc_esum_kernel(
    const float* __restrict__ s, const float* __restrict__ rp,
    float* __restrict__ esum)
{
    __shared__ float red[256];
    int i = blockIdx.x;
    int c = threadIdx.x & 127, h = threadIdx.x >> 7;
    float sic = s[(size_t)i * DD + c];
    float acc = 0.0f;
    for (int j = h * 384; j < h * 384 + 384; ++j)
        acc += fmaxf(sic + rp[(size_t)j * DD + c], 0.0f);
    red[threadIdx.x] = acc;
    __syncthreads();
    if (threadIdx.x < 128)
        atomicAdd(&esum[threadIdx.x], red[threadIdx.x] + red[threadIdx.x + 128]);
}

// out[c] = act( sum_k x[k]*xscale*W[k*128+c] + b[c] ),  1 block x 128 threads
__global__ void vecmat_kernel(const float* __restrict__ x, float xscale,
                              const float* __restrict__ W,
                              const float* __restrict__ b,
                              float* __restrict__ out, int relu_flag) {
    int c = threadIdx.x;
    float acc = b[c];
    for (int k = 0; k < DD; ++k) acc += x[k] * xscale * W[k * DD + c];
    if (relu_flag) acc = fmaxf(acc, 0.0f);
    out[c] = acc;
}

__global__ void colsum_f32_kernel(const float* __restrict__ X,
                                  float* __restrict__ out, int rows, float scale) {
    int c = threadIdx.x;
    float acc = 0.0f;
    for (int r = 0; r < rows; ++r) acc += X[(size_t)r * DD + c];
    out[c] = acc * scale;
}

__global__ void colsum_bf16_kernel(const bf16* __restrict__ X,
                                   float* __restrict__ out, int rows, float scale) {
    int c = threadIdx.x;
    float acc = 0.0f;
    for (int r = 0; r < rows; ++r) acc += (float)X[(size_t)r * DD + c];
    out[c] = acc * scale;
}

__global__ void finalize_v_kernel(const float* __restrict__ Y,
                                  const float* __restrict__ uterm,
                                  bf16* __restrict__ vout, int n) {
    int i = blockIdx.x * 256 + threadIdx.x;
    if (i < n) {
        float v = Y[i] + uterm[i & 127];
        vout[i] = (bf16)fmaxf(v, 0.0f);
    }
}

__global__ void u_update_kernel(const float* __restrict__ esum, float escale,
                                const float* __restrict__ meanv,
                                const float* __restrict__ u,
                                const float* __restrict__ We_u,
                                const float* __restrict__ Wv_u,
                                const float* __restrict__ Wu_u,
                                const float* __restrict__ bu,
                                float* __restrict__ uout) {
    int c = threadIdx.x;
    float acc = bu[c];
    for (int k = 0; k < DD; ++k) {
        acc += esum[k] * escale * We_u[k * DD + c];
        acc += meanv[k] * Wv_u[k * DD + c];
        acc += u[k] * Wu_u[k * DD + c];
    }
    uout[c] = fmaxf(acc, 0.0f);
}

__global__ void decoder_kernel(const float* __restrict__ u,
                               const float* __restrict__ W,
                               const float* __restrict__ b,
                               float* __restrict__ out) {
    __shared__ float red[128];
    int c = threadIdx.x;
    red[c] = u[c] * W[c];
    __syncthreads();
    for (int s = 64; s > 0; s >>= 1) {
        if (c < s) red[c] += red[c + s];
        __syncthreads();
    }
    if (c == 0) out[0] = red[0] + b[0];
}

// ---------------------------------------------------------------------------
// Host orchestration
// ---------------------------------------------------------------------------
extern "C" void kernel_launch(void* const* d_in, const int* in_sizes, int n_in,
                              void* d_out, int out_size, void* d_ws, size_t ws_size,
                              hipStream_t stream) {
    const float* v_in      = (const float*)d_in[0];
    const float* enc_Ws    = (const float*)d_in[1];
    const float* enc_Wr    = (const float*)d_in[2];
    const float* enc_be    = (const float*)d_in[3];
    const float* enc_Weu   = (const float*)d_in[4];
    const float* enc_bu    = (const float*)d_in[5];
    const float* core_We_e = (const float*)d_in[6];
    const float* core_Ws_e = (const float*)d_in[7];
    const float* core_Wr_e = (const float*)d_in[8];
    const float* core_Wu_e = (const float*)d_in[9];
    const float* core_be   = (const float*)d_in[10];
    const float* core_We_v = (const float*)d_in[11];
    const float* core_Wv_v = (const float*)d_in[12];
    const float* core_Wu_v = (const float*)d_in[13];
    const float* core_bv   = (const float*)d_in[14];
    const float* core_We_u = (const float*)d_in[15];
    const float* core_Wv_u = (const float*)d_in[16];
    const float* core_Wu_u = (const float*)d_in[17];
    const float* core_bu   = (const float*)d_in[18];
    const float* dec_W     = (const float*)d_in[19];
    const float* dec_b     = (const float*)d_in[20];

    char* ws = (char*)d_ws;
    size_t off = 0;
    auto carve = [&](size_t bytes) -> char* {
        char* p = ws + off;
        off += (bytes + 255) & ~(size_t)255;
        return p;
    };

    bf16* e      = (bf16*)carve((size_t)NN * NN * DD * sizeof(bf16)); // 151 MB
    bf16* wbf    = (bf16*)carve((size_t)12 * DD * DD * sizeof(bf16));
    bf16* vbf    = (bf16*)carve((size_t)NN * DD * sizeof(bf16));
    bf16* aggev  = (bf16*)carve((size_t)NN * DD * sizeof(bf16));
    float* sbuf  = (float*)carve((size_t)NN * DD * sizeof(float));
    float* rbuf  = (float*)carve((size_t)NN * DD * sizeof(float));
    float* sebuf = (float*)carve((size_t)NN * DD * sizeof(float));
    float* rebuf = (float*)carve((size_t)NN * DD * sizeof(float));
    float* Ybuf  = (float*)carve((size_t)NN * DD * sizeof(float));
    float* Sbuf  = (float*)carve((size_t)NN * DD * sizeof(float));
    float* esum  = (float*)carve(DD * sizeof(float));
    float* meanv = (float*)carve(DD * sizeof(float));
    float* uterm = (float*)carve(DD * sizeof(float));
    float* ubufA = (float*)carve(DD * sizeof(float));
    float* ubufB = (float*)carve(DD * sizeof(float));

    const int WMAT = DD * DD;  // 16384
    bf16* WencS = wbf + 0 * WMAT;
    bf16* WencR = wbf + 1 * WMAT;
    auto Wslot = [&](int k, int s) { return wbf + (2 + k * 5 + s) * WMAT; };

    // --- convert weights & node features to bf16 ---
    f32_to_bf16_kernel<<<64, 256, 0, stream>>>(enc_Ws, WencS, WMAT, 1.0f);
    f32_to_bf16_kernel<<<64, 256, 0, stream>>>(enc_Wr, WencR, WMAT, 1.0f);
    for (int k = 0; k < 2; ++k) {
        f32_to_bf16_kernel<<<64, 256, 0, stream>>>(core_Ws_e + k * WMAT, Wslot(k, 0), WMAT, 1.0f);
        f32_to_bf16_kernel<<<64, 256, 0, stream>>>(core_Wr_e + k * WMAT, Wslot(k, 1), WMAT, 1.0f);
        f32_to_bf16_kernel<<<64, 256, 0, stream>>>(core_We_e + k * WMAT, Wslot(k, 2), WMAT, 1.0f);
        f32_to_bf16_kernel<<<64, 256, 0, stream>>>(core_We_v + k * WMAT, Wslot(k, 3), WMAT, 1.0f);
        f32_to_bf16_kernel<<<64, 256, 0, stream>>>(core_Wv_v + k * WMAT, Wslot(k, 4), WMAT, 1.0f);
    }
    f32_to_bf16_kernel<<<384, 256, 0, stream>>>(v_in, vbf, NN * DD, 1.0f);

    // --- encoder: s = v@Ws, r' = v@Wr + be ; u0 = relu(mean(e0)@Weu + bu) ---
    gemm128_kernel<<<6, 256, 0, stream>>>(vbf, WencS, sbuf, 0);
    gemm128_kernel<<<6, 256, 0, stream>>>(vbf, WencR, rbuf, 0);
    add_rowvec_kernel<<<384, 256, 0, stream>>>(rbuf, enc_be, NN * DD);
    hipMemsetAsync(esum, 0, DD * sizeof(float), stream);
    enc_esum_kernel<<<NN, 256, 0, stream>>>(sbuf, rbuf, esum);
    vecmat_kernel<<<1, 128, 0, stream>>>(esum, 1.0f / (float)(NN * NN),
                                         enc_Weu, enc_bu, ubufA, 1);

    float* ucur = ubufA;
    float* unext = ubufB;

    for (int k = 0; k < 2; ++k) {
        // per-node terms for the edge update (uses old v)
        gemm128_kernel<<<6, 256, 0, stream>>>(vbf, Wslot(k, 0), sebuf, 0);
        gemm128_kernel<<<6, 256, 0, stream>>>(vbf, Wslot(k, 1), rebuf, 0);
        vecmat_kernel<<<1, 128, 0, stream>>>(ucur, 1.0f, core_Wu_e + k * WMAT,
                                             core_be + k * DD, uterm, 0);
        hipMemsetAsync(Sbuf, 0, (size_t)NN * DD * sizeof(float), stream);

        dim3 egrid(6, 48);
        if (k == 0) {
            edge_kernel<true, true><<<egrid, 256, 0, stream>>>(
                nullptr, e, sbuf, rbuf, Wslot(k, 2), sebuf, rebuf, uterm, Sbuf);
        } else {
            edge_kernel<false, false><<<egrid, 256, 0, stream>>>(
                e, nullptr, nullptr, nullptr, Wslot(k, 2), sebuf, rebuf, uterm, Sbuf);
        }

        // aggregations from the new edge state
        colsum_f32_kernel<<<1, 128, 0, stream>>>(Sbuf, esum, NN, 1.0f);
        f32_to_bf16_kernel<<<384, 256, 0, stream>>>(Sbuf, aggev, NN * DD,
                                                    1.0f / (float)NN);
        // v update (uses old v, old u)
        gemm128_kernel<<<6, 256, 0, stream>>>(aggev, Wslot(k, 3), Ybuf, 0);
        gemm128_kernel<<<6, 256, 0, stream>>>(vbf, Wslot(k, 4), Ybuf, 1);
        vecmat_kernel<<<1, 128, 0, stream>>>(ucur, 1.0f, core_Wu_v + k * WMAT,
                                             core_bv + k * DD, uterm, 0);
        finalize_v_kernel<<<384, 256, 0, stream>>>(Ybuf, uterm, vbf, NN * DD);
        // u update (uses new e, new v, old u)
        colsum_bf16_kernel<<<1, 128, 0, stream>>>(vbf, meanv, NN, 1.0f / (float)NN);
        u_update_kernel<<<1, 128, 0, stream>>>(
            esum, 1.0f / (float)(NN * NN), meanv, ucur,
            core_We_u + k * WMAT, core_Wv_u + k * WMAT, core_Wu_u + k * WMAT,
            core_bu + k * DD, unext);
        float* tmp = ucur; ucur = unext; unext = tmp;
    }

    decoder_kernel<<<1, 128, 0, stream>>>(ucur, dec_W, dec_b, (float*)d_out);
}
